// ODENet_53231824666932
// MI455X (gfx1250) — matrix operand order repaired and checked by reading, not compiled, not run on hardware
//
#include <hip/hip_runtime.h>
#include <hip/hip_bf16.h>

// ---------------------------------------------------------------------------
// Neural-ODE (dopri5, 8 fixed steps) for MI455X / gfx1250.
// Convs = implicit GEMM via v_wmma_f32_16x16x32_bf16 (bf16 A/B, f32 acc).
// Weights live resident in LDS (whole B matrix, preloaded once per block,
// via global_load_async_to_lds_b128 / ASYNCcnt); the 4KB A tile is
// double-buffered so the K-loop runs with ONE barrier per 32-K step and no
// global weight traffic.  All geometry is compile-time (no runtime division
// in device code hot paths).
// ---------------------------------------------------------------------------

#if defined(__gfx1250__) && \
    __has_builtin(__builtin_amdgcn_global_load_async_to_lds_b128) && \
    __has_builtin(__builtin_amdgcn_s_wait_asynccnt)
#define HAVE_ASYNC_LDS 1
#else
#define HAVE_ASYNC_LDS 0
#endif

typedef __attribute__((ext_vector_type(16))) __bf16 bf16x16;
typedef __attribute__((ext_vector_type(8)))  float  floatx8;
typedef int async_v4i __attribute__((vector_size(16)));   // builtin's pointee

__device__ __forceinline__ unsigned short f2bf(float f) {
    unsigned int u = __float_as_uint(f);
    unsigned int r = (u + 0x7fffu + ((u >> 16) & 1u)) >> 16;   // RNE
    return (unsigned short)r;
}
__device__ __forceinline__ float bf2f(unsigned short h) {
    return __uint_as_float(((unsigned int)h) << 16);
}

// ---------------------------------------------------------------------------
// conv1: 1->64, 3x3, stride1, pad0, 64x64 -> 62x62.  Direct conv, weights in
// LDS, one thread per output pixel computing all 64 output channels (bf16 out).
// ---------------------------------------------------------------------------
__global__ __launch_bounds__(256) void conv1_kernel(
    const float* __restrict__ x, const float* __restrict__ w,
    const float* __restrict__ bias, unsigned short* __restrict__ out)
{
    __shared__ float ws[64 * 9];
    __shared__ float bs[64];
    const int tid = threadIdx.x;
    for (int i = tid; i < 576; i += 256) ws[i] = w[i];
    if (tid < 64) bs[tid] = bias[tid];
    __syncthreads();

    const int p  = blockIdx.x * 256 + tid;      // 0 .. 256*3844-1 (exact)
    const int b  = p / 3844;
    const int r  = p - b * 3844;
    const int oy = r / 62, ox = r - oy * 62;

    const float* xb = x + (size_t)b * 4096;
    float in9[9];
#pragma unroll
    for (int ky = 0; ky < 3; ky++)
#pragma unroll
        for (int kx = 0; kx < 3; kx++)
            in9[ky * 3 + kx] = xb[(oy + ky) * 64 + (ox + kx)];

    for (int oc = 0; oc < 64; oc++) {
        float a = bs[oc];
#pragma unroll
        for (int k = 0; k < 9; k++) a += ws[oc * 9 + k] * in9[k];
        out[((size_t)b * 64 + oc) * 3844 + r] = f2bf(a);
    }
}

// ---------------------------------------------------------------------------
// Weight repack: OIHW f32 -> TRANSPOSED [N=64][Kpad] bf16, zero-padded K.
// K index = c*KH*KW + kh*KW + kw.   wb[n*Kpad + k]
// ---------------------------------------------------------------------------
__global__ void repack_kernel(const float* __restrict__ w,
                              unsigned short* __restrict__ wb,
                              int Cin, int KHW, int Kpad)
{
    int idx = blockIdx.x * 256 + threadIdx.x;
    if (idx >= Kpad * 64) return;
    int n = idx / Kpad;
    int k = idx - n * Kpad;
    int Kreal = Cin * KHW;
    unsigned short v = 0;
    if (k < Kreal) {
        int c = k / KHW, r = k - c * KHW;
        v = f2bf(w[((size_t)n * Cin + c) * KHW + r]);
    }
    wb[idx] = v;
}

// ---------------------------------------------------------------------------
// GroupNorm, one block per (batch, group); group = 2 contiguous channels.
// MODE 0: bf16 in/out (may alias), +ReLU      (gn1, gn2, f_gn2)
// MODE 1: f32 in, bf16 out, +ReLU             (f_gn1)
// MODE 2: f32 in, f32 out, no ReLU            (f_gn3 -> k_i)
// ---------------------------------------------------------------------------
template <int MODE>
__global__ __launch_bounds__(256) void gn_kernel(
    const void* __restrict__ inp, void* __restrict__ outp,
    const float* __restrict__ gamma, const float* __restrict__ beta, int HW)
{
    const int b = blockIdx.x >> 5, g = blockIdx.x & 31;
    const int c0 = g * 2;
    const int n  = 2 * HW;
    const size_t base = ((size_t)b * 64 + c0) * (size_t)HW;
    const float* inf          = (const float*)inp;
    const unsigned short* inh = (const unsigned short*)inp;

    float s = 0.f, s2 = 0.f;
    for (int i = threadIdx.x; i < n; i += 256) {
        float v = (MODE == 0) ? bf2f(inh[base + i]) : inf[base + i];
        s += v; s2 += v * v;
    }
    __shared__ float r1[256], r2[256];
    r1[threadIdx.x] = s; r2[threadIdx.x] = s2;
    __syncthreads();
    for (int o = 128; o > 0; o >>= 1) {
        if (threadIdx.x < o) { r1[threadIdx.x] += r1[threadIdx.x + o];
                               r2[threadIdx.x] += r2[threadIdx.x + o]; }
        __syncthreads();
    }
    const float mean = r1[0] / (float)n;
    const float var  = r2[0] / (float)n - mean * mean;
    const float rstd = rsqrtf(var + 1e-5f);

    for (int i = threadIdx.x; i < n; i += 256) {
        int c = c0 + (i >= HW);
        float v = (MODE == 0) ? bf2f(inh[base + i]) : inf[base + i];
        float xn = (v - mean) * rstd * gamma[c] + beta[c];
        if (MODE != 2) xn = fmaxf(xn, 0.f);
        if (MODE == 2) ((float*)outp)[base + i] = xn;
        else           ((unsigned short*)outp)[base + i] = f2bf(xn);
    }
}

// ---------------------------------------------------------------------------
// Implicit-GEMM conv via WMMA bf16, fully compile-time geometry.
//   A = im2col patches [M = 256*OH*OW][K], built on the fly into LDS
//       (HAS_T synthesizes the leading t-channel of the concat).
//   B = transposed weights [64][KPAD], RESIDENT in LDS for the whole kernel.
// Block: 256 thr = 8 wave32; tile M=64 x N=64.  Wave w: msub = w&3 (16 A
// rows), nhalf = w>>2 (32 B cols) -> 2 WMMAs per K-step reusing the A frag.
// A tile double-buffered: stage kt+1 while kt computes -> 1 barrier/step.
// A and B use contiguous K order (lane half h holds k = 16h..16h+15) so
// every fragment is one contiguous 32-byte LDS read (ISA 7.12.2).
// ---------------------------------------------------------------------------
template <int KH, int KW, int STRIDE, int PAD, bool HAS_T, bool OUT_F32,
          int H, int W, int OH, int OW, int KREAL, int KTILES>
__global__ __launch_bounds__(256) void conv_wmma_kernel(
    const unsigned short* __restrict__ in,   // bf16 activations [256,64,H,W]
    const unsigned short* __restrict__ wb,   // bf16 weights [64][KPAD]
    const float* __restrict__ bias, float tval,
    unsigned short* __restrict__ outh, float* __restrict__ outf)
{
    constexpr int KHW  = KH * KW;
    constexpr int PPI  = OH * OW;
    constexpr int TOT  = 256 * PPI;
    constexpr int CHW  = H * W;
    constexpr int KPAD = KTILES * 32;

    __shared__ alignas(32) unsigned short a_sh[2 * 64 * 32];   // double buffer
    __shared__ alignas(32) unsigned short b_sh[64 * KPAD];     // resident B

    const int tid   = threadIdx.x;
    const int lane  = tid & 31, wv = tid >> 5;
    const int msub  = wv & 3, nhalf = wv >> 2;
    const int pix0  = blockIdx.x * 64;
    const unsigned short tb = f2bf(tval);

    // ---- loop-invariant A-staging state: this thread writes rows
    //      mi = (tid>>5) + e*8 at K column kk = tid&31 every K-step ----
    const int kk    = tid & 31;
    const int aphys = kk;
    int  abase[8], aoff[8], oys[8], oxs[8];
    bool mok[8];
#pragma unroll
    for (int e = 0; e < 8; e++) {
        int mi = (tid >> 5) + e * 8;
        int p  = pix0 + mi;
        mok[e] = p < TOT;
        int pc = mok[e] ? p : 0;
        int img = pc / PPI, rp = pc - img * PPI;     // compile-time divisor
        int oy = rp / OW,  ox = rp - oy * OW;
        abase[e] = img * 64 * CHW;
        oys[e]   = oy * STRIDE - PAD;
        oxs[e]   = ox * STRIDE - PAD;
        aoff[e]  = mi * 32 + aphys;
    }

    auto stageA = [&](int kt, unsigned short* dst) {
        const int  k   = kt * 32 + kk;
        const int  kc  = k / KHW;                    // compile-time divisors
        const int  rr  = k - kc * KHW;
        const int  kh  = rr / KW, kw2 = rr - kh * KW;
        const bool kok = (k < KREAL);
        const int  cc  = HAS_T ? (kc - 1) : kc;
#pragma unroll
        for (int e = 0; e < 8; e++) {
            int  iy  = oys[e] + kh, ix = oxs[e] + kw2;
            bool inb = kok & mok[e] & (iy >= 0) & (iy < H) & (ix >= 0) & (ix < W);
            bool ld  = inb && (!HAS_T || kc != 0);
            int  off = ld ? (abase[e] + cc * CHW + iy * W + ix) : 0;
            unsigned short raw = in[off];
            unsigned short v = ld ? raw
                : (unsigned short)((HAS_T && inb && kc == 0) ? tb
                                                             : (unsigned short)0);
            dst[aoff[e]] = v;
        }
    };

    // ---- preload the entire B matrix into LDS (once per block) ----
#if HAVE_ASYNC_LDS
    for (int i = tid; i < KPAD * 8; i += 256) {
        unsigned short* wsrc = const_cast<unsigned short*>(wb) + (size_t)i * 8;
        __builtin_amdgcn_global_load_async_to_lds_b128(
            (__attribute__((address_space(1))) async_v4i*)wsrc,
            (__attribute__((address_space(3))) async_v4i*)(b_sh + (size_t)i * 8),
            0, 0);
    }
#else
    for (int i = tid; i < KPAD * 8; i += 256)
        *(uint4*)(b_sh + (size_t)i * 8) = *(const uint4*)(wb + (size_t)i * 8);
#endif
    stageA(0, a_sh);
#if HAVE_ASYNC_LDS
    __builtin_amdgcn_s_wait_asynccnt(0);
#endif
    __syncthreads();

    floatx8 acc0 = {}, acc1 = {};
    const int h = lane >> 4;

    for (int kt = 0; kt < KTILES; ++kt) {
        const unsigned short* cur = a_sh + (kt & 1) * (64 * 32);
        bf16x16 af = *reinterpret_cast<const bf16x16*>(
            cur + (msub * 16 + (lane & 15)) * 32 + h * 16);
        bf16x16 b0 = *reinterpret_cast<const bf16x16*>(
            b_sh + (nhalf * 32 + (lane & 15)) * KPAD + kt * 32 + h * 16);
        bf16x16 b1 = *reinterpret_cast<const bf16x16*>(
            b_sh + (nhalf * 32 + 16 + (lane & 15)) * KPAD + kt * 32 + h * 16);
        acc0 = __builtin_amdgcn_wmma_f32_16x16x32_bf16(
            false, af, false, b0, (short)0, acc0, false, false);
        acc1 = __builtin_amdgcn_wmma_f32_16x16x32_bf16(
            false, af, false, b1, (short)0, acc1, false, false);
        if (kt + 1 < KTILES)
            stageA(kt + 1, a_sh + ((kt + 1) & 1) * (64 * 32));
        __syncthreads();
    }

    // ---- epilogue: +bias, store (D: n = lane%16, m = (lane/16)*8 + r) ----
#pragma unroll
    for (int half = 0; half < 2; half++) {
        const int nn = nhalf * 32 + half * 16 + (lane & 15);
        const float bv = bias[nn];
#pragma unroll
        for (int r = 0; r < 8; r++) {
            int m = msub * 16 + (lane >> 4) * 8 + r;
            int p = pix0 + m;
            if (p < TOT) {
                int img = p / PPI, rp = p - img * PPI;
                float val = (half ? acc1[r] : acc0[r]) + bv;
                size_t oidx = ((size_t)img * 64 + nn) * PPI + rp;
                if (OUT_F32) outf[oidx] = val;
                else         outh[oidx] = f2bf(val);
            }
        }
    }
}

// ---------------------------------------------------------------------------
// dopri5 combo: dst = y + hs * sum(c_i * k_i), zero coeffs skip the load.
// ---------------------------------------------------------------------------
__global__ __launch_bounds__(256) void combo_kernel(
    float* __restrict__ dst, const float* __restrict__ y,
    const float* __restrict__ k0, const float* __restrict__ k1,
    const float* __restrict__ k2, const float* __restrict__ k3,
    const float* __restrict__ k4, const float* __restrict__ k5,
    float c0, float c1, float c2, float c3, float c4, float c5,
    float hs, int n)
{
    int i = blockIdx.x * 256 + threadIdx.x;
    if (i >= n) return;
    float a = 0.f;
    if (c0 != 0.f) a += c0 * k0[i];
    if (c1 != 0.f) a += c1 * k1[i];
    if (c2 != 0.f) a += c2 * k2[i];
    if (c3 != 0.f) a += c3 * k3[i];
    if (c4 != 0.f) a += c4 * k4[i];
    if (c5 != 0.f) a += c5 * k5[i];
    dst[i] = y[i] + hs * a;
}

// ---------------------------------------------------------------------------
// out GN + ReLU + global average pool (15x15). One block per (batch, group).
// ---------------------------------------------------------------------------
__global__ __launch_bounds__(256) void gn_pool_kernel(
    const float* __restrict__ y, const float* __restrict__ gamma,
    const float* __restrict__ beta, float* __restrict__ pooled)
{
    const int b = blockIdx.x >> 5, g = blockIdx.x & 31;
    const int c0 = g * 2;
    const size_t base = ((size_t)b * 64 + c0) * 225;
    const int n = 450;

    float s = 0.f, s2 = 0.f;
    for (int i = threadIdx.x; i < n; i += 256) {
        float v = y[base + i]; s += v; s2 += v * v;
    }
    __shared__ float r1[256], r2[256];
    r1[threadIdx.x] = s; r2[threadIdx.x] = s2;
    __syncthreads();
    for (int o = 128; o > 0; o >>= 1) {
        if (threadIdx.x < o) { r1[threadIdx.x] += r1[threadIdx.x + o];
                               r2[threadIdx.x] += r2[threadIdx.x + o]; }
        __syncthreads();
    }
    const float mean = r1[0] / (float)n;
    const float rstd = rsqrtf(r2[0] / (float)n - mean * mean + 1e-5f);
    __syncthreads();

    float p0 = 0.f, p1 = 0.f;
    for (int i = threadIdx.x; i < n; i += 256) {
        int c = c0 + (i >= 225);
        float xn = fmaxf((y[base + i] - mean) * rstd * gamma[c] + beta[c], 0.f);
        if (i < 225) p0 += xn; else p1 += xn;
    }
    r1[threadIdx.x] = p0; r2[threadIdx.x] = p1;
    __syncthreads();
    for (int o = 128; o > 0; o >>= 1) {
        if (threadIdx.x < o) { r1[threadIdx.x] += r1[threadIdx.x + o];
                               r2[threadIdx.x] += r2[threadIdx.x + o]; }
        __syncthreads();
    }
    if (threadIdx.x == 0) {
        pooled[(size_t)b * 64 + c0]     = r1[0] / 225.f;
        pooled[(size_t)b * 64 + c0 + 1] = r2[0] / 225.f;
    }
}

// ---------------------------------------------------------------------------
// FC: [256,64] @ [10,64]^T + b -> [256,10]
// ---------------------------------------------------------------------------
__global__ __launch_bounds__(256) void fc_kernel(
    const float* __restrict__ pooled, const float* __restrict__ w,
    const float* __restrict__ b, float* __restrict__ out)
{
    int i = blockIdx.x * 256 + threadIdx.x;
    if (i >= 2560) return;
    int bi = i / 10, j = i - bi * 10;
    float a = b[j];
#pragma unroll 8
    for (int c = 0; c < 64; c++) a += pooled[bi * 64 + c] * w[j * 64 + c];
    out[i] = a;
}

// ---------------------------------------------------------------------------
extern "C" void kernel_launch(void* const* d_in, const int* in_sizes, int n_in,
                              void* d_out, int out_size, void* d_ws, size_t ws_size,
                              hipStream_t stream)
{
    (void)in_sizes; (void)n_in; (void)out_size; (void)ws_size;

    const float* x        = (const float*)d_in[0];
    const float* conv1_w  = (const float*)d_in[1];
    const float* conv1_b  = (const float*)d_in[2];
    const float* gn1_g    = (const float*)d_in[3];
    const float* gn1_b    = (const float*)d_in[4];
    const float* conv2_w  = (const float*)d_in[5];
    const float* conv2_b  = (const float*)d_in[6];
    const float* gn2_g    = (const float*)d_in[7];
    const float* gn2_b    = (const float*)d_in[8];
    const float* conv3_w  = (const float*)d_in[9];
    const float* conv3_b  = (const float*)d_in[10];
    const float* fgn1_g   = (const float*)d_in[11];
    const float* fgn1_b   = (const float*)d_in[12];
    const float* fconv1_w = (const float*)d_in[13];
    const float* fconv1_b = (const float*)d_in[14];
    const float* fgn2_g   = (const float*)d_in[15];
    const float* fgn2_b   = (const float*)d_in[16];
    const float* fconv2_w = (const float*)d_in[17];
    const float* fconv2_b = (const float*)d_in[18];
    const float* fgn3_g   = (const float*)d_in[19];
    const float* fgn3_b   = (const float*)d_in[20];
    const float* ogn_g    = (const float*)d_in[21];
    const float* ogn_b    = (const float*)d_in[22];
    const float* fc_w     = (const float*)d_in[23];
    const float* fc_b     = (const float*)d_in[24];

    // ---- workspace layout (~194 MB peak; phase-A t0 region reused by ODE) --
    char* ws = (char*)d_ws;
    const size_t MB = 1024u * 1024u;
    float* y   = (float*)(ws + 0);                 // ODE state  [256,64,15,15]
    float* yc  = (float*)(ws + 16 * MB);           // stage input
    float* kb[6];
    for (int i = 0; i < 6; i++) kb[i] = (float*)(ws + (32 + 16 * (size_t)i) * MB);
    unsigned short* abf   = (unsigned short*)(ws + 128 * MB);  // bf16 f_gn1 out
    unsigned short* abf2  = (unsigned short*)(ws + 136 * MB);  // bf16 f_conv1 out
    float*          otmp  = (float*)(ws + 144 * MB);           // f32 f_conv2 out
    unsigned short* t0buf = (unsigned short*)(ws + 0);         // phase-A only
    unsigned short* t1buf = (unsigned short*)(ws + 160 * MB);
    unsigned short* wb2   = (unsigned short*)(ws + 192 * MB);
    unsigned short* wb3   = (unsigned short*)(ws + 192 * MB + 256 * 1024);
    unsigned short* wbf1  = (unsigned short*)(ws + 192 * MB + 512 * 1024);
    unsigned short* wbf2  = (unsigned short*)(ws + 192 * MB + 768 * 1024);
    float* pooled         = (float*)(ws + 193 * MB);
    float* outp           = (float*)d_out;

    const int NY = 256 * 64 * 225;                 // 3,686,400

    // ---- weight repack to bf16 [N=64][Kpad] (transposed for resident-B) ----
    repack_kernel<<<256, 256, 0, stream>>>(conv2_w,  wb2,  64, 16, 1024);
    repack_kernel<<<256, 256, 0, stream>>>(conv3_w,  wb3,  64, 16, 1024);
    repack_kernel<<<152, 256, 0, stream>>>(fconv1_w, wbf1, 65,  9,  608);
    repack_kernel<<<152, 256, 0, stream>>>(fconv2_w, wbf2, 65,  9,  608);

    // ---- downsampling path ----
    conv1_kernel<<<3844, 256, 0, stream>>>(x, conv1_w, conv1_b, t0buf);
    gn_kernel<0><<<8192, 256, 0, stream>>>(t0buf, t0buf, gn1_g, gn1_b, 3844);
    conv_wmma_kernel<4, 4, 2, 1, false, false, 62, 62, 31, 31, 1024, 32>
        <<<3844, 256, 0, stream>>>(t0buf, wb2, conv2_b, 0.f,
                                   t1buf, (float*)nullptr);
    gn_kernel<0><<<8192, 256, 0, stream>>>(t1buf, t1buf, gn2_g, gn2_b, 961);
    conv_wmma_kernel<4, 4, 2, 1, false, true, 31, 31, 15, 15, 1024, 32>
        <<<900, 256, 0, stream>>>(t1buf, wb3, conv3_b, 0.f,
                                  (unsigned short*)nullptr, y);

    // ---- ODE func f(t, y) ----
    auto evalf = [&](const float* src, float t, float* kout) {
        gn_kernel<1><<<8192, 256, 0, stream>>>(src, abf, fgn1_g, fgn1_b, 225);
        conv_wmma_kernel<3, 3, 1, 1, true, false, 15, 15, 15, 15, 585, 19>
            <<<900, 256, 0, stream>>>(abf, wbf1, fconv1_b, t,
                                      abf2, (float*)nullptr);
        gn_kernel<0><<<8192, 256, 0, stream>>>(abf2, abf2, fgn2_g, fgn2_b, 225);
        conv_wmma_kernel<3, 3, 1, 1, true, true, 15, 15, 15, 15, 585, 19>
            <<<900, 256, 0, stream>>>(abf2, wbf2, fconv2_b, t,
                                      (unsigned short*)nullptr, otmp);
        gn_kernel<2><<<8192, 256, 0, stream>>>(otmp, kout, fgn3_g, fgn3_b, 225);
    };

    // ---- dopri5 fixed-step integration ----
    const float hs = 1.f / 8.f;
    const float CO[5][6] = {
        {1.f/5, 0, 0, 0, 0, 0},
        {3.f/40, 9.f/40, 0, 0, 0, 0},
        {44.f/45, -56.f/15, 32.f/9, 0, 0, 0},
        {19372.f/6561, -25360.f/2187, 64448.f/6561, -212.f/729, 0, 0},
        {9017.f/3168, -355.f/33, 46732.f/5247, 49.f/176, -5103.f/18656, 0},
    };
    const float FI[6] = {35.f/384, 0.f, 500.f/1113, 125.f/192,
                         -2187.f/6784, 11.f/84};
    const float TOFF[6] = {0.f, 0.2f, 0.3f, 0.8f, 8.f/9.f, 1.f};

    for (int s = 0; s < 8; s++) {
        float tbase = (float)s * hs;
        evalf(y, tbase, kb[0]);
        for (int j = 1; j < 6; j++) {
            combo_kernel<<<14400, 256, 0, stream>>>(
                yc, y, kb[0], kb[1], kb[2], kb[3], kb[4], kb[5],
                CO[j-1][0], CO[j-1][1], CO[j-1][2],
                CO[j-1][3], CO[j-1][4], CO[j-1][5], hs, NY);
            evalf(yc, tbase + TOFF[j] * hs, kb[j]);
        }
        combo_kernel<<<14400, 256, 0, stream>>>(
            y, y, kb[0], kb[1], kb[2], kb[3], kb[4], kb[5],
            FI[0], FI[1], FI[2], FI[3], FI[4], FI[5], hs, NY);
    }

    // ---- head ----
    gn_pool_kernel<<<8192, 256, 0, stream>>>(y, ogn_g, ogn_b, pooled);
    fc_kernel<<<10, 256, 0, stream>>>(pooled, fc_w, fc_b, outp);
}
